// EdgeMaskGenerator_8916352106738
// MI455X (gfx1250) — compile-verified, matrix operand order checked by reading, not run on hardware
//
#include <hip/hip_runtime.h>
#include <hip/hip_bf16.h>
#include <math.h>

typedef __bf16 bf16;
typedef __attribute__((ext_vector_type(16))) __bf16 v16bf;
typedef __attribute__((ext_vector_type(8)))  __bf16 v8bf;
typedef __attribute__((ext_vector_type(8)))  float  v8f;

#define NF   256   // F (node feature dim)
#define K2F  512   // 2F (concat feature dim, GEMM K)
#define NH   256   // H (hidden dim, GEMM N)
#define TPB  256   // threads per block (8 waves)

__global__ __launch_bounds__(TPB)
void edge_mlp_wmma_kernel(const float* __restrict__ x,
                          const long long* __restrict__ edge_index,
                          const float* __restrict__ W1,
                          const float* __restrict__ b1,
                          const float* __restrict__ W2,
                          const float* __restrict__ b2,
                          float* __restrict__ out,
                          int E)
{
    // W1 (H x 2F) staged as bf16 row-major: 256*512*2 = 256 KB of LDS (<= 320 KB/WGP)
    __shared__ __align__(16) bf16 sW1[NH * K2F];

    const int tid = threadIdx.x;

    // ---- Stage W1 f32 -> bf16 into LDS (vectorized: float4 in, 8B out) ----
    {
        const float4* src = (const float4*)W1;
        const int n4 = (NH * K2F) / 4;            // 32768 float4 groups
        for (int i = tid; i < n4; i += TPB) {
            float4 v = src[i];
            union { bf16 h[4]; uint2 u; } pk;
            pk.h[0] = (bf16)v.x; pk.h[1] = (bf16)v.y;
            pk.h[2] = (bf16)v.z; pk.h[3] = (bf16)v.w;
            ((uint2*)sW1)[i] = pk.u;
        }
    }
    __syncthreads();

    const int lane  = tid & 31;
    const int wave  = tid >> 5;
    const int m     = lane & 15;   // edge row (M) within the 16-edge tile
    const int khalf = lane >> 4;   // selects K sub-groups per the 16-bit A layout

    const long long tile = (long long)blockIdx.x * (TPB / 32) + wave;
    const long long eb   = tile * 16;
    if (eb >= E) return;

    long long e = eb + m; if (e >= E) e = E - 1;
    const int nrow = (int)edge_index[e];
    const int ncol = (int)edge_index[(long long)E + e];
    const float* __restrict__ rowp = x + (long long)nrow * NF;
    const float* __restrict__ colp = x + (long long)ncol * NF;

    // ---- Build all 16 A-fragments (16 edges x 512 K, bf16) in registers ----
    // ISA 16-bit A 16x32 layout: lanes 0-15 -> elems 0..7 = K0..7, 8..15 = K16..23
    //                            lanes 16-31 -> elems 0..7 = K8..15, 8..15 = K24..31
    v16bf a[16];
    #pragma unroll
    for (int kt = 0; kt < 16; ++kt) {
        const int kb = kt * 32;
        const float* src = (kb < NF) ? rowp : colp;
        const int f0 = ((kb < NF) ? kb : (kb - NF)) + khalf * 8;
        const int f1 = f0 + 16;
        float4 p0 = *(const float4*)(src + f0);
        float4 p1 = *(const float4*)(src + f0 + 4);
        float4 p2 = *(const float4*)(src + f1);
        float4 p3 = *(const float4*)(src + f1 + 4);
        v16bf av;
        av[0]  = (bf16)p0.x; av[1]  = (bf16)p0.y; av[2]  = (bf16)p0.z; av[3]  = (bf16)p0.w;
        av[4]  = (bf16)p1.x; av[5]  = (bf16)p1.y; av[6]  = (bf16)p1.z; av[7]  = (bf16)p1.w;
        av[8]  = (bf16)p2.x; av[9]  = (bf16)p2.y; av[10] = (bf16)p2.z; av[11] = (bf16)p2.w;
        av[12] = (bf16)p3.x; av[13] = (bf16)p3.y; av[14] = (bf16)p3.z; av[15] = (bf16)p3.w;
        a[kt] = av;
    }

    // ---- Sweep 16 N-tiles of the hidden layer; fuse bias+ReLU+W2 dot ----
    float p[8];
    #pragma unroll
    for (int i = 0; i < 8; ++i) p[i] = 0.0f;

    #pragma unroll 1
    for (int nt = 0; nt < 16; ++nt) {
        const int n = nt * 16 + m;                       // this lane's N column
        const bf16* __restrict__ wr = sW1 + (long long)n * K2F;
        v8f c = {};
        #pragma unroll
        for (int kt = 0; kt < 16; ++kt) {
            const int kb = kt * 32;
            const int g0 = kb + khalf * 8;
            const int g1 = g0 + 16;
            uint4 q0 = *(const uint4*)(wr + g0);         // 8 bf16 (16B, aligned)
            uint4 q1 = *(const uint4*)(wr + g1);
            v8bf b0 = __builtin_bit_cast(v8bf, q0);
            v8bf b1v = __builtin_bit_cast(v8bf, q1);
            v16bf bv;
            #pragma unroll
            for (int j = 0; j < 8; ++j) { bv[j] = b0[j]; bv[8 + j] = b1v[j]; }
            c = __builtin_amdgcn_wmma_f32_16x16x32_bf16(
                    false, a[kt], false, bv, (short)0, c, false, false);
        }
        const float b1n = b1[n];
        const float w2n = W2[n];
        #pragma unroll
        for (int i = 0; i < 8; ++i) {
            float h = c[i] + b1n;                        // C layout: VGPR i -> M=i / i+8
            h = h > 0.0f ? h : 0.0f;
            p[i] += h * w2n;
        }
    }

    // ---- Reduce partial dot over N within each 16-lane half ----
    #pragma unroll
    for (int i = 0; i < 8; ++i) {
        p[i] += __shfl_xor(p[i], 8, 32);
        p[i] += __shfl_xor(p[i], 4, 32);
        p[i] += __shfl_xor(p[i], 2, 32);
        p[i] += __shfl_xor(p[i], 1, 32);
    }

    // lanes 0-15 hold edges eb+0..7 in p[0..7]; lanes 16-31 hold eb+8..15
    if (m == 0) {
        const float bias2 = b2[0];
        const long long base = eb + (long long)khalf * 8;
        #pragma unroll
        for (int i = 0; i < 8; ++i) {
            const long long eo = base + i;
            if (eo < E) {
                const float logit = p[i] + bias2;
                out[eo] = 1.0f / (1.0f + __expf(-logit));
            }
        }
    }
}

extern "C" void kernel_launch(void* const* d_in, const int* in_sizes, int n_in,
                              void* d_out, int out_size, void* d_ws, size_t ws_size,
                              hipStream_t stream) {
    const float*     x  = (const float*)d_in[0];
    const long long* ei = (const long long*)d_in[1];   // int64 edge_index (2,E)
    const float*     W1 = (const float*)d_in[2];
    const float*     b1 = (const float*)d_in[3];
    const float*     W2 = (const float*)d_in[4];
    const float*     b2 = (const float*)d_in[5];
    float*           out = (float*)d_out;

    const int E = in_sizes[1] / 2;
    const int tilesPerBlock = TPB / 32;                 // 8 waves, 16 edges each
    const int nTiles  = (E + 15) / 16;
    const int nBlocks = (nTiles + tilesPerBlock - 1) / tilesPerBlock;

    edge_mlp_wmma_kernel<<<nBlocks, TPB, 0, stream>>>(x, ei, W1, b1, W2, b2, out, E);
}
